// LCGN_BACKBONE_9440338116920
// MI455X (gfx1250) — matrix operand order repaired and checked by reading, not compiled, not run on hardware
//
#include <hip/hip_runtime.h>
#include <stdint.h>
#include <math.h>

typedef __bf16 bf16_t;
typedef __attribute__((ext_vector_type(16))) __bf16 v16bf;
typedef __attribute__((ext_vector_type(8)))  __bf16 bf16x8;
typedef __attribute__((ext_vector_type(8)))  float  v8f;

typedef union { v16bf v; bf16x8 h[2]; } fragU;

#define B_   32
#define N_   1024
#define LQ_  128
#define DF_  2112
#define CTX_ 512
#define CMD_ 512
#define ROWS (B_*N_)          // 32768
#define NEGF (-1e30f)

__device__ __forceinline__ bf16_t f2bf(float x){ return (bf16_t)x; }

// ---------------------------------------------------------------------------
// bf16 WMMA GEMM:  C[M,N] = alpha * A[M,K] @ B + bias,  B given TRANSPOSED:
//   Bt stored [N][K] row-major (ldb >= K), logical B[k][n] = Bt[n*ldb + k].
// Optional f32 C (ldc) and/or bf16 D16 (ldd). batch via grid.z.
// Block tile 128x128, 8 waves (4x2), wave tile 32x64 (2x4 WMMA 16x16x32 bf16).
// Double-buffered LDS; A and B staged in contiguous k order so each lane's
// fragment is 32 contiguous bytes (2x ds_load_b128, no movs).
// ---------------------------------------------------------------------------
#define BM 128
#define BN 128
#define BK 32
#define LDP (BK + 8)          // 40 bf16 = 80B row stride (16B mult, conflict-free)

__global__ __launch_bounds__(256)
void k_wmma_gemm(const bf16_t* __restrict__ A, long long lda, long long sA,
                 const bf16_t* __restrict__ Bt, long long ldb, long long sB,
                 float* __restrict__ C, long long ldc, long long sC,
                 bf16_t* __restrict__ D16, long long ldd, long long sD,
                 const float* __restrict__ bias,
                 float alpha, int M, int N, int K)
{
    __shared__ __align__(16) bf16_t As[2][BM][LDP];
    __shared__ __align__(16) bf16_t Bs[2][BN][LDP];

    const int bz = blockIdx.z;
    A  += (long long)bz * sA;
    Bt += (long long)bz * sB;
    if (C)   C   += (long long)bz * sC;
    if (D16) D16 += (long long)bz * sD;

    const int tid  = threadIdx.x;
    const int lane = tid & 31;
    const int wave = tid >> 5;       // 0..7
    const int wm   = wave >> 1;      // 0..3 (M)
    const int wn   = wave & 1;       // 0..1 (N)
    const int rowBase = blockIdx.x * BM;
    const int colBase = blockIdx.y * BN;
    const int mlo = lane & 15;
    const int hse = lane >> 4;

    // staging indices (fixed per thread): 2 chunks each for A and B
    const int ar  = tid >> 2;                 // A tile rows ar, ar+64
    const int kcA = (tid & 3) * 8;            // global k offset within tile
    const int kpA = kcA;  // LDS pos (contiguous k, same as B)
    const int bn  = tid >> 2;                 // B tile rows (n) bn, bn+64
    const int kcB = (tid & 3) * 8;

    const bool allFull = (rowBase + BM <= M) && (colBase + BN <= N);
    const bool pa0 = (rowBase + ar      < M);
    const bool pa1 = (rowBase + ar + 64 < M);
    const bool pb0 = (colBase + bn      < N);
    const bool pb1 = (colBase + bn + 64 < N);

    const bf16_t* aP0 = A  + (long long)(rowBase + ar) * lda + kcA;
    const bf16_t* aP1 = aP0 + 64LL * lda;
    const bf16_t* bP0 = Bt + (long long)(colBase + bn) * ldb + kcB;
    const bf16_t* bP1 = bP0 + 64LL * ldb;

    bf16x8 z8;
    #pragma unroll
    for (int i = 0; i < 8; ++i) z8[i] = f2bf(0.f);

    bf16x8 pa[2], pb[2];
    auto fetch = [&](long long koff) {
        if (allFull) {
            pa[0] = *(const bf16x8*)(aP0 + koff);
            pa[1] = *(const bf16x8*)(aP1 + koff);
            pb[0] = *(const bf16x8*)(bP0 + koff);
            pb[1] = *(const bf16x8*)(bP1 + koff);
        } else {
            pa[0] = pa0 ? *(const bf16x8*)(aP0 + koff) : z8;
            pa[1] = pa1 ? *(const bf16x8*)(aP1 + koff) : z8;
            pb[0] = pb0 ? *(const bf16x8*)(bP0 + koff) : z8;
            pb[1] = pb1 ? *(const bf16x8*)(bP1 + koff) : z8;
        }
    };
    auto stage = [&](int buf) {
        *(bf16x8*)&As[buf][ar     ][kpA] = pa[0];
        *(bf16x8*)&As[buf][ar + 64][kpA] = pa[1];
        *(bf16x8*)&Bs[buf][bn     ][kcB] = pb[0];
        *(bf16x8*)&Bs[buf][bn + 64][kcB] = pb[1];
    };

    v8f zero8 = {0.f,0.f,0.f,0.f,0.f,0.f,0.f,0.f};
    v8f acc[2][4];
    #pragma unroll
    for (int i = 0; i < 2; ++i)
        #pragma unroll
        for (int j = 0; j < 4; ++j) acc[i][j] = zero8;

    const int NT = K / BK;
    fetch(0);
    stage(0);
    __syncthreads();

    for (int kt = 0; kt < NT; ++kt) {
        const int cur = kt & 1;
        if (kt + 1 < NT) fetch((long long)(kt + 1) * BK);

        fragU af[2];
        #pragma unroll
        for (int i = 0; i < 2; ++i) {
            int m = wm * 32 + i * 16 + mlo;
            af[i].h[0] = *(const bf16x8*)&As[cur][m][hse * 16];
            af[i].h[1] = *(const bf16x8*)&As[cur][m][hse * 16 + 8];
        }
        fragU bf[4];
        #pragma unroll
        for (int j = 0; j < 4; ++j) {
            int n = wn * 64 + j * 16 + mlo;
            bf[j].h[0] = *(const bf16x8*)&Bs[cur][n][hse * 16];
            bf[j].h[1] = *(const bf16x8*)&Bs[cur][n][hse * 16 + 8];
        }
        #pragma unroll
        for (int i = 0; i < 2; ++i)
            #pragma unroll
            for (int j = 0; j < 4; ++j)
                acc[i][j] = __builtin_amdgcn_wmma_f32_16x16x32_bf16(
                    false, af[i].v, false, bf[j].v, (short)0, acc[i][j], false, false);

        if (kt + 1 < NT) stage(cur ^ 1);
        __syncthreads();
    }

    // ---- epilogue ----
    #pragma unroll
    for (int i = 0; i < 2; ++i) {
        #pragma unroll
        for (int j = 0; j < 4; ++j) {
            int ncol = colBase + wn * 64 + j * 16 + mlo;
            bool okn = allFull || (ncol < N);
            float bv = (bias && okn) ? bias[ncol] : 0.f;
            #pragma unroll
            for (int r = 0; r < 8; ++r) {
                int mrow = rowBase + wm * 32 + i * 16 + hse * 8 + r;
                if (allFull || (mrow < M && okn)) {
                    float v = acc[i][j][r] * alpha + bv;
                    if (C)   C[(long long)mrow * ldc + ncol]   = v;
                    if (D16) D16[(long long)mrow * ldd + ncol] = f2bf(v);
                }
            }
        }
    }
}

// ---------------------------------------------------------------------------
// Elementwise / reduction kernels
// ---------------------------------------------------------------------------
__global__ void k_f2bf(const float* __restrict__ in, bf16_t* __restrict__ out, long long n) {
    long long i = (long long)blockIdx.x * blockDim.x + threadIdx.x;
    if (i < n) out[i] = f2bf(in[i]);
}

// f32 W[K][N] -> bf16 Wt[N][K]   (one-time, small)
__global__ void k_f2bf_T(const float* __restrict__ in, bf16_t* __restrict__ out,
                         int K, int N) {
    long long i = (long long)blockIdx.x * 256 + threadIdx.x;   // output index
    if (i < (long long)K * N) {
        int n = (int)(i / K);
        int k = (int)(i % K);
        out[i] = f2bf(in[(long long)k * N + n]);
    }
}

__global__ __launch_bounds__(256)
void k_normalize(const float* __restrict__ img, bf16_t* __restrict__ out) {
    __shared__ float sm[256];
    int row = blockIdx.x;
    const float* r = img + (size_t)row * DF_;
    float s = 0.f;
    for (int d = threadIdx.x; d < DF_; d += 256) { float v = r[d]; s += v * v; }
    sm[threadIdx.x] = s; __syncthreads();
    for (int o = 128; o > 0; o >>= 1) { if (threadIdx.x < o) sm[threadIdx.x] += sm[threadIdx.x + o]; __syncthreads(); }
    float scale = 1.f / fmaxf(sqrtf(sm[0]), 1e-12f);
    bf16_t* o = out + (size_t)row * DF_;
    for (int d = threadIdx.x; d < DF_; d += 256) o[d] = f2bf(r[d] * scale);
}

__global__ void k_fill_ctx(const float* __restrict__ initMem, bf16_t* __restrict__ xj) {
    long long i = (long long)blockIdx.x * 256 + threadIdx.x;   // ROWS*512
    long long r = i >> 9; int d = (int)(i & 511);
    xj[r * 1536 + 512 + d] = f2bf(initMem[d]);
}

__global__ void k_elu_bf(const float* __restrict__ in, bf16_t* __restrict__ out, int n) {
    int i = blockIdx.x * 256 + threadIdx.x;
    if (i < n) { float x = in[i]; out[i] = f2bf(x > 0.f ? x : (expf(x) - 1.f)); }
}

__global__ __launch_bounds__(256)
void k_text_raw(const float* __restrict__ qcmd, const float* __restrict__ lstm,
                const float* __restrict__ c2lw, const float* __restrict__ c2lb,
                float* __restrict__ raw) {
    __shared__ float sm[256];
    int bl = blockIdx.x; int b = bl >> 7; int l = bl & 127;
    const float* q = qcmd + b * CMD_;
    const float* x = lstm + ((size_t)b * LQ_ + l) * CMD_;
    float s = 0.f;
    for (int d = threadIdx.x; d < CMD_; d += 256) s += q[d] * x[d] * c2lw[d];
    sm[threadIdx.x] = s; __syncthreads();
    for (int o = 128; o > 0; o >>= 1) { if (threadIdx.x < o) sm[threadIdx.x] += sm[threadIdx.x + o]; __syncthreads(); }
    if (threadIdx.x == 0) raw[bl] = sm[0] + c2lb[0];
}

__global__ __launch_bounds__(128)
void k_text_softmax(const float* __restrict__ raw, const int* __restrict__ qlen,
                    float* __restrict__ att) {
    __shared__ float sm[128];
    int b = blockIdx.x; int l = threadIdx.x;
    float v = (l >= qlen[b]) ? NEGF : raw[b * LQ_ + l];
    sm[l] = v; __syncthreads();
    for (int o = 64; o > 0; o >>= 1) { if (l < o) sm[l] = fmaxf(sm[l], sm[l + o]); __syncthreads(); }
    float mx = sm[0]; __syncthreads();
    float e = expf(v - mx);
    sm[l] = e; __syncthreads();
    for (int o = 64; o > 0; o >>= 1) { if (l < o) sm[l] += sm[l + o]; __syncthreads(); }
    att[b * LQ_ + l] = e / sm[0];
}

__global__ __launch_bounds__(256)
void k_text_cmd(const float* __restrict__ att, const float* __restrict__ lstm,
                bf16_t* __restrict__ cmd16) {
    int b = blockIdx.x;
    for (int d = threadIdx.x; d < CMD_; d += 256) {
        float s = 0.f;
        for (int l = 0; l < LQ_; ++l) s += att[b * LQ_ + l] * lstm[((size_t)b * LQ_ + l) * CMD_ + d];
        cmd16[b * CMD_ + d] = f2bf(s);
    }
}

__global__ void k_joint_prod(const float* __restrict__ pl, const float* __restrict__ pc,
                             bf16_t* __restrict__ xj) {
    long long i = (long long)blockIdx.x * 256 + threadIdx.x;   // ROWS*512
    long long r = i >> 9; int d = (int)(i & 511);
    xj[r * 1536 + 1024 + d] = f2bf(pl[i] * pc[i]);
}

// k-gate: out[m][d] = in[m][d] * gate[b][d]   (keeps [N][K] layout for q@k^T)
__global__ void k_gate_mul(const float* __restrict__ in, const float* __restrict__ gate,
                           bf16_t* __restrict__ out) {
    long long i = (long long)blockIdx.x * 256 + threadIdx.x;   // ROWS*512
    int b = (int)(i >> 19);
    int d = (int)(i & 511);
    out[i] = f2bf(in[i] * gate[(b << 9) + d]);
}

// v-gate + transpose: out[b][d][m] = in[b*1024+m][d] * gate[b][d]
// tiled 64x64 through LDS so both global accesses stay coalesced.
__global__ __launch_bounds__(256)
void k_gate_mul_T(const float* __restrict__ in, const float* __restrict__ gate,
                  bf16_t* __restrict__ out) {
    __shared__ bf16_t t[64][65];
    int mt = blockIdx.x;                 // 0..511  (ROWS/64)
    int dt = blockIdx.y;                 // 0..7    (512/64)
    int b    = (mt * 64) >> 10;
    int mloc = (mt * 64) & 1023;
    #pragma unroll
    for (int p = 0; p < 16; ++p) {
        int e = p * 256 + threadIdx.x;
        int r = e >> 6, c = e & 63;
        float v = in[((long long)mt * 64 + r) * 512 + dt * 64 + c] * gate[b * 512 + dt * 64 + c];
        t[r][c] = f2bf(v);
    }
    __syncthreads();
    #pragma unroll
    for (int p = 0; p < 16; ++p) {
        int e = p * 256 + threadIdx.x;
        int dr = e >> 6, mc = e & 63;
        out[((long long)b * 512 + dt * 64 + dr) * 1024 + mloc + mc] = t[mc][dr];
    }
}

__global__ void k_copy_ctx(const bf16_t* __restrict__ xj, bf16_t* __restrict__ cat) {
    long long i = (long long)blockIdx.x * 256 + threadIdx.x;   // ROWS*512
    long long r = i >> 9; int d = (int)(i & 511);
    cat[r * 1024 + d] = xj[r * 1536 + 512 + d];
}

__global__ __launch_bounds__(256)
void k_attn_softmax(const float* __restrict__ score, const int* __restrict__ entn,
                    bf16_t* __restrict__ prob) {
    __shared__ float sm[256];
    int blk = blockIdx.x;            // b*1024 + n
    int b = blk >> 10; int n = blk & 1023;
    int ent = entn[b];
    const float* srow = score + (size_t)blk * N_;
    bf16_t* prow = prob + (size_t)blk * N_;
    bool rowm = (n >= ent);
    float v[4]; float mx = NEGF;
    #pragma unroll
    for (int j = 0; j < 4; ++j) {
        int m = threadIdx.x + j * 256;
        float x = (rowm || m >= ent) ? NEGF : srow[m];
        v[j] = x; mx = fmaxf(mx, x);
    }
    sm[threadIdx.x] = mx; __syncthreads();
    for (int o = 128; o > 0; o >>= 1) { if (threadIdx.x < o) sm[threadIdx.x] = fmaxf(sm[threadIdx.x], sm[threadIdx.x + o]); __syncthreads(); }
    mx = sm[0]; __syncthreads();
    float s = 0.f;
    #pragma unroll
    for (int j = 0; j < 4; ++j) { v[j] = expf(v[j] - mx); s += v[j]; }
    sm[threadIdx.x] = s; __syncthreads();
    for (int o = 128; o > 0; o >>= 1) { if (threadIdx.x < o) sm[threadIdx.x] += sm[threadIdx.x + o]; __syncthreads(); }
    float inv = 1.f / sm[0];
    #pragma unroll
    for (int j = 0; j < 4; ++j) prow[threadIdx.x + j * 256] = f2bf(v[j] * inv);
}

// ---------------------------------------------------------------------------
extern "C" void kernel_launch(void* const* d_in, const int* in_sizes, int n_in,
                              void* d_out, int out_size, void* d_ws, size_t ws_size,
                              hipStream_t stream)
{
    (void)in_sizes; (void)n_in; (void)out_size; (void)ws_size;

    const float* images   = (const float*)d_in[0];
    const float* q_enc    = (const float*)d_in[1];
    const float* lstm     = (const float*)d_in[2];
    const int*   q_len    = (const int*)  d_in[3];
    const int*   ent_n    = (const int*)  d_in[4];
    const float* initKB_w = (const float*)d_in[5];
    const float* initKB_b = (const float*)d_in[6];
    const float* initMem  = (const float*)d_in[7];
    const float* qIn_w    = (const float*)d_in[8];
    const float* qIn_b    = (const float*)d_in[9];
    const float* qIn2_w   = (const float*)d_in[10];
    const float* qIn2_b   = (const float*)d_in[11];
    const float* c2l_w    = (const float*)d_in[12];
    const float* c2l_b    = (const float*)d_in[13];
    const float* pxl_w    = (const float*)d_in[14]; const float* pxl_b = (const float*)d_in[15];
    const float* pxc_w    = (const float*)d_in[16]; const float* pxc_b = (const float*)d_in[17];
    const float* qry_w    = (const float*)d_in[18]; const float* qry_b = (const float*)d_in[19];
    const float* key_w    = (const float*)d_in[20]; const float* key_b = (const float*)d_in[21];
    const float* val_w    = (const float*)d_in[22]; const float* val_b = (const float*)d_in[23];
    const float* pk_w     = (const float*)d_in[24]; const float* pk_b  = (const float*)d_in[25];
    const float* pv_w     = (const float*)d_in[26]; const float* pv_b  = (const float*)d_in[27];
    const float* mu_w     = (const float*)d_in[28]; const float* mu_b  = (const float*)d_in[29];
    const float* ck_w     = (const float*)d_in[30]; const float* ck_b  = (const float*)d_in[31];
    float* out = (float*)d_out;

    char* cur = (char*)d_ws;
    auto alloc = [&](size_t bytes) -> char* {
        char* p = cur; cur += (bytes + 255) & ~(size_t)255; return p;
    };

    bf16_t* imgs16 = (bf16_t*)alloc((size_t)ROWS * DF_ * 2);
    bf16_t* xjoint = (bf16_t*)alloc((size_t)ROWS * 1536 * 2);   // [x_loc | x_ctx | prod]
    bf16_t* q16    = (bf16_t*)alloc((size_t)ROWS * 512 * 2);
    bf16_t* k16    = (bf16_t*)alloc((size_t)ROWS * 512 * 2);
    bf16_t* v16t   = (bf16_t*)alloc((size_t)ROWS * 512 * 2);    // [b][d][m] transposed
    bf16_t* prob16 = (bf16_t*)alloc((size_t)B_ * N_ * N_ * 2);
    bf16_t* cat16  = (bf16_t*)alloc((size_t)ROWS * 1024 * 2);   // [x_ctx_old | message]
    float*  S1     = (float*) alloc((size_t)ROWS * 512 * 4);
    float*  S2     = (float*) alloc((size_t)ROWS * 512 * 4);
    float*  scoreF = (float*) alloc((size_t)B_ * N_ * N_ * 4);  // also hosts v-raw f32

    // transposed bf16 weights: Wt[N][K]
    bf16_t* wInitKB = (bf16_t*)alloc((size_t)DF_ * 512 * 2);
    bf16_t* wQin    = (bf16_t*)alloc((size_t)512 * 512 * 2);
    bf16_t* wQin2   = (bf16_t*)alloc((size_t)4 * 512 * 512 * 2);
    bf16_t* wPxl    = (bf16_t*)alloc((size_t)512 * 512 * 2);
    bf16_t* wPxc    = (bf16_t*)alloc((size_t)512 * 512 * 2);
    bf16_t* wQry    = (bf16_t*)alloc((size_t)1536 * 512 * 2);
    bf16_t* wKey    = (bf16_t*)alloc((size_t)1536 * 512 * 2);
    bf16_t* wVal    = (bf16_t*)alloc((size_t)1536 * 512 * 2);
    bf16_t* wPk     = (bf16_t*)alloc((size_t)512 * 512 * 2);
    bf16_t* wPv     = (bf16_t*)alloc((size_t)512 * 512 * 2);
    bf16_t* wMu     = (bf16_t*)alloc((size_t)1024 * 512 * 2);
    bf16_t* wCk     = (bf16_t*)alloc((size_t)1024 * 512 * 2);
    bf16_t* qenc16  = (bf16_t*)alloc((size_t)32 * 512 * 2);
    bf16_t* qbase16 = (bf16_t*)alloc((size_t)32 * 512 * 2);
    bf16_t* cmd16   = (bf16_t*)alloc((size_t)32 * 512 * 2);
    float*  qtmpF   = (float*) alloc((size_t)32 * 512 * 4);
    float*  qcmdF   = (float*) alloc((size_t)32 * 512 * 4);
    float*  rawF    = (float*) alloc((size_t)32 * 128 * 4);
    float*  attF    = (float*) alloc((size_t)32 * 128 * 4);
    float*  pkgF    = (float*) alloc((size_t)32 * 512 * 4);
    float*  pvgF    = (float*) alloc((size_t)32 * 512 * 4);

    auto convT = [&](const float* src, bf16_t* dst, int K, int N) {
        long long n = (long long)K * N;
        k_f2bf_T<<<dim3((unsigned)((n + 255) / 256)), dim3(256), 0, stream>>>(src, dst, K, N);
    };
    auto gemm = [&](const bf16_t* A, long long lda, long long sA,
                    const bf16_t* Bt, long long ldb, long long sB,
                    float* C, long long ldc, long long sC,
                    bf16_t* D, long long ldd, long long sD,
                    const float* bias, float alpha, int M, int Nn, int K, int batch) {
        dim3 g((unsigned)((M + BM - 1) / BM), (unsigned)((Nn + BN - 1) / BN), (unsigned)batch);
        k_wmma_gemm<<<g, dim3(256), 0, stream>>>(A, lda, sA, Bt, ldb, sB,
                                                 C, ldc, sC, D, ldd, sD, bias, alpha, M, Nn, K);
    };

    const unsigned GB512 = (unsigned)((size_t)ROWS * 512 / 256);  // 65536 blocks

    // ---- one-time transposed conversions ----
    convT(initKB_w, wInitKB, DF_, 512);
    convT(qIn_w,  wQin,  512, 512);
    for (int t = 0; t < 4; ++t)
        convT(qIn2_w + (size_t)t * 512 * 512, wQin2 + (size_t)t * 512 * 512, 512, 512);
    convT(pxl_w, wPxl, 512, 512);
    convT(pxc_w, wPxc, 512, 512);
    convT(qry_w, wQry, 1536, 512);
    convT(key_w, wKey, 1536, 512);
    convT(val_w, wVal, 1536, 512);
    convT(pk_w,  wPk,  512, 512);
    convT(pv_w,  wPv,  512, 512);
    convT(mu_w,  wMu,  1024, 512);
    convT(ck_w,  wCk,  1024, 512);
    k_f2bf<<<dim3(64), dim3(256), 0, stream>>>(q_enc, qenc16, 16384);

    // ---- x_loc = normalize(images) @ initKB_w + b -> xjoint[:,0:512] ----
    k_normalize<<<dim3(ROWS), dim3(256), 0, stream>>>(images, imgs16);
    gemm(imgs16, DF_, 0, wInitKB, DF_, 0,
         nullptr, 0, 0, xjoint, 1536, 0, initKB_b, 1.f, ROWS, 512, DF_, 1);

    // ---- x_ctx init -> xjoint[:,512:1024] ----
    k_fill_ctx<<<dim3(GB512), dim3(256), 0, stream>>>(initMem, xjoint);

    // ---- q_base = elu(q_enc @ qIn_w + b) ----
    gemm(qenc16, 512, 0, wQin, 512, 0, qtmpF, 512, 0, nullptr, 0, 0, qIn_b, 1.f, 32, 512, 512, 1);
    k_elu_bf<<<dim3(64), dim3(256), 0, stream>>>(qtmpF, qbase16, 16384);

    const float scale = 0.044194173824159216f;  // 1/sqrt(512)

    for (int t = 0; t < 4; ++t) {
        // ---- textual command ----
        gemm(qbase16, 512, 0, wQin2 + (size_t)t * 512 * 512, 512, 0,
             qcmdF, 512, 0, nullptr, 0, 0, qIn2_b + t * 512, 1.f, 32, 512, 512, 1);
        k_text_raw<<<dim3(B_ * LQ_), dim3(256), 0, stream>>>(qcmdF, lstm, c2l_w, c2l_b, rawF);
        k_text_softmax<<<dim3(B_), dim3(128), 0, stream>>>(rawF, q_len, attF);
        k_text_cmd<<<dim3(B_), dim3(256), 0, stream>>>(attF, lstm, cmd16);
        gemm(cmd16, 512, 0, wPk, 512, 0, pkgF, 512, 0, nullptr, 0, 0, pk_b, 1.f, 32, 512, 512, 1);
        gemm(cmd16, 512, 0, wPv, 512, 0, pvgF, 512, 0, nullptr, 0, 0, pv_b, 1.f, 32, 512, 512, 1);

        // ---- proj_loc, proj_ctx; xjoint[:,1024:1536] = prod ----
        gemm(xjoint,       1536, 0, wPxl, 512, 0, S1, 512, 0, nullptr, 0, 0, pxl_b, 1.f, ROWS, 512, 512, 1);
        gemm(xjoint + 512, 1536, 0, wPxc, 512, 0, S2, 512, 0, nullptr, 0, 0, pxc_b, 1.f, ROWS, 512, 512, 1);
        k_joint_prod<<<dim3(GB512), dim3(256), 0, stream>>>(S1, S2, xjoint);

        // ---- q, k, v ----
        gemm(xjoint, 1536, 0, wQry, 1536, 0, nullptr, 0, 0, q16, 512, 0, qry_b, 1.f, ROWS, 512, 1536, 1);
        gemm(xjoint, 1536, 0, wKey, 1536, 0, S2, 512, 0, nullptr, 0, 0, key_b, 1.f, ROWS, 512, 1536, 1);
        k_gate_mul<<<dim3(GB512), dim3(256), 0, stream>>>(S2, pkgF, k16);
        gemm(xjoint, 1536, 0, wVal, 1536, 0, scoreF, 512, 0, nullptr, 0, 0, val_b, 1.f, ROWS, 512, 1536, 1);
        k_gate_mul_T<<<dim3(512, 8), dim3(256), 0, stream>>>(scoreF, pvgF, v16t);

        // ---- score = (q @ k^T) * scale  (batched; k16 is already [N][K]) ----
        gemm(q16, 512, (long long)N_ * 512, k16, 512, (long long)N_ * 512,
             scoreF, N_, (long long)N_ * N_, nullptr, 0, 0, nullptr, scale, N_, N_, 512, B_);
        k_attn_softmax<<<dim3(ROWS), dim3(256), 0, stream>>>(scoreF, ent_n, prob16);

        // ---- cat = [x_ctx_old | message]; message = prob @ v ----
        k_copy_ctx<<<dim3(GB512), dim3(256), 0, stream>>>(xjoint, cat16);
        gemm(prob16, N_, (long long)N_ * N_, v16t, 1024, 512LL * 1024,
             nullptr, 0, 0, cat16 + 512, 1024, (long long)N_ * 1024, nullptr, 1.f, N_, 512, N_, B_);

        // ---- x_ctx = cat @ mu_w + b -> xjoint[:,512:1024] ----
        gemm(cat16, 1024, 0, wMu, 1024, 0, nullptr, 0, 0, xjoint + 512, 1536, 0,
             mu_b, 1.f, ROWS, 512, 1024, 1);
    }

    // ---- out = [x_loc | x_ctx] @ ck_w + b  (A = xjoint cols 0:1024) ----
    gemm(xjoint, 1536, 0, wCk, 1024, 0, out, 512, 0, nullptr, 0, 0,
         ck_b, 1.f, ROWS, 512, 1024, 1);
}
